// Attention_68358699483637
// MI455X (gfx1250) — matrix operand order repaired and checked by reading, not compiled, not run on hardware
//
#include <hip/hip_runtime.h>

#define HIDDEN 2048
#define SEQ    2048
#define NH     16
#define NKV    4
#define HD     128
#define QKV_COLS 3072   // 2048 Q | 512 K | 512 V
#define KOFF   2048
#define VOFF   2560

typedef __bf16 bf16;
typedef __attribute__((ext_vector_type(16))) __bf16 v16bf;
typedef __attribute__((ext_vector_type(8)))  __bf16 v8bf;
typedef __attribute__((ext_vector_type(4)))  __bf16 v4bf;
typedef __attribute__((ext_vector_type(8)))  float  floatx8;
typedef __attribute__((ext_vector_type(4)))  float  floatx4;

#if __has_builtin(__builtin_amdgcn_global_load_async_to_lds_b128)
#define ASYNC_LDS_COPY 1
typedef int v4i_vs __attribute__((vector_size(16)));
typedef __attribute__((address_space(1))) v4i_vs* gptr_v4i;
typedef __attribute__((address_space(3))) v4i_vs* lptr_v4i;
static __device__ __forceinline__ void async_copy16(const void* g, void* l) {
  __builtin_amdgcn_global_load_async_to_lds_b128(
      (gptr_v4i)(void*)(size_t)(const char*)g,
      (lptr_v4i)l, 0, 0);
}
static __device__ __forceinline__ void async_wait0() {
#if __has_builtin(__builtin_amdgcn_s_wait_asynccnt)
  __builtin_amdgcn_s_wait_asynccnt(0);
#else
  __asm__ volatile("s_wait_asynccnt 0" ::: "memory");
#endif
}
#endif

static __device__ __forceinline__ floatx8 wmma_bf16(v16bf a, v16bf b, floatx8 c) {
  return __builtin_amdgcn_wmma_f32_16x16x32_bf16(false, a, false, b, (short)0, c, false, false);
}

static __device__ __forceinline__ v16bf cat8(v8bf lo, v8bf hi) {
  return __builtin_shufflevector(lo, hi, 0,1,2,3,4,5,6,7,8,9,10,11,12,13,14,15);
}

// A fragment: 16x32 (MxK) from row-major bf16, row stride ld (elems, 16B-aligned
// byte stride). ISA: lanes 0-15: M=lane, K={0..7,16..23}; lanes 16-31: K={8..15,24..31}.
static __device__ __forceinline__ v16bf load_frag_a(const bf16* p, int ld) {
  int lane = threadIdx.x & 31;
  const bf16* r = p + (lane & 15) * ld + ((lane >> 4) << 3);
  return cat8(*(const v8bf*)r, *(const v8bf*)(r + 16));
}

// B fragment: 32x16 (KxN) from a K-transposed tile T[n][k], row stride ld.
// ISA: lanes 0-15: N=lane, K={0..7,16..23}; lanes 16-31: N=lane-16, K={8..15,24..31}.
static __device__ __forceinline__ v16bf load_frag_bT(const bf16* p, int ld) {
  int lane = threadIdx.x & 31;
  const bf16* r = p + (lane & 15) * ld + ((lane >> 4) << 3);
  return cat8(*(const v8bf*)r, *(const v8bf*)(r + 16));
}

// ---------------------------------------------------------------------------
// Tiled bf16-WMMA GEMM: Y[2048 x (c0+c1+c2)] = X[2048 x K] * [W0|W1|W2]
// block = 128 threads (4 waves), tile 128x64, K-step 32.
// Per wave per K-step: 2 A frags + 4 shared B frags -> 8 WMMAs.
// TA = float (convert) or bf16 (pure copy, async if available). TY = bf16/float.
// ---------------------------------------------------------------------------
#define XS_LD 40   // 80B rows: 16B aligned, 20-bank stride
#define WS_LD 40

template <typename TA, typename TY>
__global__ void __launch_bounds__(128)
gemm_bf16_kernel(const TA* __restrict__ X,
                 const float* __restrict__ W0,
                 const float* __restrict__ W1,
                 const float* __restrict__ W2,
                 TY* __restrict__ Y,
                 int K, int c0, int c1, int c2) {
  const int NC = c0 + c1 + c2;
  __shared__ __align__(16) bf16 Xs[128][XS_LD];
  __shared__ __align__(16) bf16 WsT[64][WS_LD];   // [n][k]

  int n0 = blockIdx.x * 64;
  int m0 = blockIdx.y * 128;

  const float* W; int wn, wcols;
  if (n0 < c0)           { W = W0; wn = n0;            wcols = c0; }
  else if (n0 < c0 + c1) { W = W1; wn = n0 - c0;       wcols = c1; }
  else                   { W = W2; wn = n0 - c0 - c1;  wcols = c2; }

  int t    = threadIdx.x;
  int wave = t >> 5;
  int lane = t & 31;

  floatx8 z = {0.f,0.f,0.f,0.f,0.f,0.f,0.f,0.f};
  floatx8 acc[2][4] = {{z, z, z, z}, {z, z, z, z}};

  for (int k0 = 0; k0 < K; k0 += 32) {
    // A tile 128x32
    if constexpr (__is_same(TA, float)) {
#pragma unroll
      for (int r = 0; r < 8; ++r) {
        int idx = (t + r * 128) * 4;
        int row = idx >> 5, col = idx & 31;
        floatx4 f = *(const floatx4*)&X[(size_t)(m0 + row) * K + k0 + col];
        v4bf b;
#pragma unroll
        for (int i = 0; i < 4; ++i) b[i] = (bf16)f[i];
        *(v4bf*)&Xs[row][col] = b;
      }
    } else {
#pragma unroll
      for (int r = 0; r < 4; ++r) {
        int idx = (t + r * 128) * 8;
        int row = idx >> 5, col = idx & 31;
#ifdef ASYNC_LDS_COPY
        async_copy16(&X[(size_t)(m0 + row) * K + k0 + col], &Xs[row][col]);
#else
        *(v8bf*)&Xs[row][col] = *(const v8bf*)&X[(size_t)(m0 + row) * K + k0 + col];
#endif
      }
    }
    // W tile 32x64, stored transposed: coalesced float4 read, b16 scatter store
#pragma unroll
    for (int r = 0; r < 4; ++r) {
      int idx = (t + r * 128) * 4;
      int row = idx >> 6, col = idx & 63;     // row=k, col=n
      floatx4 f = *(const floatx4*)&W[(size_t)(k0 + row) * wcols + wn + col];
#pragma unroll
      for (int i = 0; i < 4; ++i) WsT[col + i][row] = (bf16)f[i];
    }
#ifdef ASYNC_LDS_COPY
    if constexpr (!__is_same(TA, float)) async_wait0();
#endif
    __syncthreads();

    v16bf a0 = load_frag_a(&Xs[wave * 16][0],      XS_LD);
    v16bf a1 = load_frag_a(&Xs[64 + wave * 16][0], XS_LD);
    v16bf b[4];
#pragma unroll
    for (int j = 0; j < 4; ++j) b[j] = load_frag_bT(&WsT[j * 16][0], WS_LD);
#pragma unroll
    for (int j = 0; j < 4; ++j) acc[0][j] = wmma_bf16(a0, b[j], acc[0][j]);
#pragma unroll
    for (int j = 0; j < 4; ++j) acc[1][j] = wmma_bf16(a1, b[j], acc[1][j]);
    __syncthreads();
  }

  int cn = lane & 15;
  int mh = (lane >> 4) * 8;
#pragma unroll
  for (int s2 = 0; s2 < 2; ++s2) {
    int base = m0 + s2 * 64 + wave * 16 + mh;
#pragma unroll
    for (int j = 0; j < 4; ++j) {
#pragma unroll
      for (int v = 0; v < 8; ++v)
        Y[(size_t)(base + v) * NC + n0 + j * 16 + cn] = (TY)acc[s2][j][v];
    }
  }
}

// ---------------------------------------------------------------------------
// Per-(token, head) RMSNorm + RoPE on Q (heads 0..15) and K (heads 16..19),
// in place in the bf16 QKV workspace; math in f32. block = 128 = head_dim.
// ---------------------------------------------------------------------------
__global__ void __launch_bounds__(128)
norm_rope_kernel(bf16* __restrict__ QKV,
                 const float* __restrict__ cosb,
                 const float* __restrict__ sinb,
                 const float* __restrict__ qscale,
                 const float* __restrict__ kscale) {
  int s = blockIdx.x;
  int h = blockIdx.y;                 // 0..19
  int d = threadIdx.x;                // 0..127

  bf16* row;
  const float* sc;
  if (h < NH) { row = QKV + (size_t)s * QKV_COLS + h * HD;                 sc = qscale; }
  else        { row = QKV + (size_t)s * QKV_COLS + KOFF + (h - NH) * HD;   sc = kscale; }

  float x  = (float)row[d];
  float ss = x * x;
#pragma unroll
  for (int off = 16; off > 0; off >>= 1) ss += __shfl_xor(ss, off, 32);

  __shared__ float wss[4];
  __shared__ float xn[HD];
  if ((d & 31) == 0) wss[d >> 5] = ss;
  __syncthreads();
  float tot = wss[0] + wss[1] + wss[2] + wss[3];
  float inv = rsqrtf(tot * (1.0f / HD) + 1e-6f);
  float y   = x * inv * sc[d];
  xn[d] = y;
  __syncthreads();
  float rot = (d < 64) ? -xn[d + 64] : xn[d - 64];
  row[d] = (bf16)(y * cosb[(size_t)s * HD + d] + rot * sinb[(size_t)s * HD + d]);
}

// ---------------------------------------------------------------------------
// Causal GQA flash attention over the bf16 workspace. block = 128 (4 waves);
// each wave owns 16 query rows; block covers 64 rows of one head. KV chunk 32.
// K staged row-major via async global->LDS copy; V staged transposed.
// ---------------------------------------------------------------------------
#define KS_LD 136   // 272B rows: 16B aligned, conflict-free frag reads
#define VS_LD 40
#define PS_LD 40

__global__ void __launch_bounds__(128)
attn_kernel(const bf16* __restrict__ QKV, bf16* __restrict__ O) {
  int h  = blockIdx.y;                 // 0..15
  int r0 = blockIdx.x * 64;
  int kvh = h >> 2;
  const bf16* Kp = QKV + KOFF + kvh * HD;
  const bf16* Vp = QKV + VOFF + kvh * HD;

  __shared__ __align__(16) bf16 Ks[32][KS_LD];     // [kv][dim]
  __shared__ __align__(16) bf16 VsT[HD][VS_LD];    // [dim][kv]
  __shared__ __align__(16) bf16 Ps[4][16][PS_LD];  // per-wave P staging

  int t    = threadIdx.x;
  int wave = t >> 5;
  int lane = t & 31;
  int cn   = lane & 15;
  int mh   = (lane >> 4) * 8;

  // Q fragments (16 rows x 128 dims): direct global b128 loads, bf16.
  v16bf qf[4];
  {
    int kb = (lane >> 4) << 3;
    const bf16* qr = QKV + (size_t)(r0 + wave * 16 + (lane & 15)) * QKV_COLS + h * HD;
#pragma unroll
    for (int c = 0; c < 4; ++c)
      qf[c] = cat8(*(const v8bf*)(qr + c * 32 + kb),
                   *(const v8bf*)(qr + c * 32 + 16 + kb));
  }

  floatx8 z = {0.f,0.f,0.f,0.f,0.f,0.f,0.f,0.f};
  floatx8 oacc[8] = {z, z, z, z, z, z, z, z};
  float rm[8], rl[8];
#pragma unroll
  for (int v = 0; v < 8; ++v) { rm[v] = -1e30f; rl[v] = 0.f; }

  const float scale = 0.08838834764831845f;   // 1/sqrt(128)
  int nchunks = (r0 + 64) / 32;

  for (int c = 0; c < nchunks; ++c) {
    int kv0 = c * 32;
    // Stage K (async bf16 copy) and V (manual transpose), 32x128 each.
#pragma unroll
    for (int r = 0; r < 4; ++r) {
      int idx = (t + r * 128) * 8;
      int row = idx >> 7, col = idx & 127;    // row=kv, col=dim
#ifdef ASYNC_LDS_COPY
      async_copy16(&Kp[(size_t)(kv0 + row) * QKV_COLS + col], &Ks[row][col]);
#else
      *(v8bf*)&Ks[row][col] = *(const v8bf*)&Kp[(size_t)(kv0 + row) * QKV_COLS + col];
#endif
      v8bf fv = *(const v8bf*)&Vp[(size_t)(kv0 + row) * QKV_COLS + col];
#pragma unroll
      for (int i = 0; i < 8; ++i) VsT[col + i][row] = fv[i];
    }
#ifdef ASYNC_LDS_COPY
    async_wait0();
#endif
    __syncthreads();

    // S = Q * K^T (16x32 as two 16x16 subtiles); Ks[n][k] is already T.
    floatx8 s0 = z, s1 = z;
#pragma unroll
    for (int kc = 0; kc < 4; ++kc) {
      v16bf b0 = load_frag_bT(&Ks[0][kc * 32],  KS_LD);
      v16bf b1 = load_frag_bT(&Ks[16][kc * 32], KS_LD);
      s0 = wmma_bf16(qf[kc], b0, s0);
      s1 = wmma_bf16(qf[kc], b1, s1);
    }

    // Scale + causal mask (C layout: VGPR v -> row mh+v, lane -> col cn).
    float sv0[8], sv1[8];
#pragma unroll
    for (int v = 0; v < 8; ++v) {
      int qrow = r0 + wave * 16 + mh + v;
      sv0[v] = (kv0 + cn      <= qrow) ? s0[v] * scale : -1e30f;
      sv1[v] = (kv0 + 16 + cn <= qrow) ? s1[v] * scale : -1e30f;
    }

    // Online softmax; row stats reduce across the 16-lane half.
    float corr[8];
#pragma unroll
    for (int v = 0; v < 8; ++v) {
      float mx = fmaxf(sv0[v], sv1[v]);
#pragma unroll
      for (int off = 8; off > 0; off >>= 1) mx = fmaxf(mx, __shfl_xor(mx, off, 32));
      float mnew = fmaxf(rm[v], mx);
      float p0 = __expf(sv0[v] - mnew);
      float p1 = __expf(sv1[v] - mnew);
      float rs = p0 + p1;
#pragma unroll
      for (int off = 8; off > 0; off >>= 1) rs += __shfl_xor(rs, off, 32);
      float cf = __expf(rm[v] - mnew);
      rl[v] = rl[v] * cf + rs;
      rm[v] = mnew;
      corr[v] = cf;
      sv0[v] = p0; sv1[v] = p1;
    }
#pragma unroll
    for (int j = 0; j < 8; ++j)
#pragma unroll
      for (int v = 0; v < 8; ++v) oacc[j][v] *= corr[v];

    // Re-layout P (C layout -> A fragment) through per-wave LDS.
#pragma unroll
    for (int v = 0; v < 8; ++v) {
      Ps[wave][mh + v][cn]      = (bf16)sv0[v];
      Ps[wave][mh + v][16 + cn] = (bf16)sv1[v];
    }
    __asm__ volatile("s_wait_dscnt 0" ::: "memory");   // wave-local LDS RAW
    v16bf pf = load_frag_a(&Ps[wave][0][0], PS_LD);

    // O += P * V (16x32 * 32x128); VsT[n][k] gives contiguous B frags.
#pragma unroll
    for (int j = 0; j < 8; ++j) {
      v16bf vb = load_frag_bT(&VsT[j * 16][0], VS_LD);
      oacc[j] = wmma_bf16(pf, vb, oacc[j]);
    }
    __syncthreads();   // before next chunk overwrites Ks/VsT
  }

#pragma unroll
  for (int j = 0; j < 8; ++j) {
#pragma unroll
    for (int v = 0; v < 8; ++v) {
      int row = r0 + wave * 16 + mh + v;
      int col = h * HD + j * 16 + cn;
      O[(size_t)row * HIDDEN + col] = (bf16)(oacc[j][v] / rl[v]);
    }
  }
}

// ---------------------------------------------------------------------------
// Final RMSNorm, in place on d_out (f32). block = 256 threads per token row.
// ---------------------------------------------------------------------------
__global__ void __launch_bounds__(256)
final_norm_kernel(float* __restrict__ Y, const float* __restrict__ scale) {
  int s = blockIdx.x;
  int t = threadIdx.x;
  float* row = Y + (size_t)s * HIDDEN;
  float vals[8];
  float ss = 0.f;
#pragma unroll
  for (int i = 0; i < 8; ++i) { vals[i] = row[t + i * 256]; ss += vals[i] * vals[i]; }
#pragma unroll
  for (int off = 16; off > 0; off >>= 1) ss += __shfl_xor(ss, off, 32);
  __shared__ float wss[8];
  if ((t & 31) == 0) wss[t >> 5] = ss;
  __syncthreads();
  float tot = 0.f;
#pragma unroll
  for (int i = 0; i < 8; ++i) tot += wss[i];
  float inv = rsqrtf(tot * (1.0f / HIDDEN) + 1e-6f);
#pragma unroll
  for (int i = 0; i < 8; ++i) row[t + i * 256] = vals[i] * inv * scale[t + i * 256];
}

extern "C" void kernel_launch(void* const* d_in, const int* in_sizes, int n_in,
                              void* d_out, int out_size, void* d_ws, size_t ws_size,
                              hipStream_t stream) {
  const float* X    = (const float*)d_in[0];   // hidden_states [1,2048,2048]
  const float* cosb = (const float*)d_in[1];   // [2048,1,128]
  const float* sinb = (const float*)d_in[2];
  const float* Wq   = (const float*)d_in[3];   // [2048,2048]
  const float* Wk   = (const float*)d_in[4];   // [2048,512]
  const float* Wv   = (const float*)d_in[5];   // [2048,512]
  const float* Wo   = (const float*)d_in[6];   // [2048,2048]
  const float* qs   = (const float*)d_in[7];
  const float* ks   = (const float*)d_in[8];
  const float* ls   = (const float*)d_in[9];
  // d_in[10] = attention_mask (causal, handled analytically)

  float* out  = (float*)d_out;
  bf16* qkv   = (bf16*)d_ws;                                  // [2048,3072] bf16
  bf16* attO  = qkv + (size_t)SEQ * QKV_COLS;                 // [2048,2048] bf16

  // 1) QKV projection (f32 in, bf16 out)
  gemm_bf16_kernel<float, bf16><<<dim3(QKV_COLS / 64, SEQ / 128), dim3(128), 0, stream>>>(
      X, Wq, Wk, Wv, qkv, HIDDEN, HIDDEN, 512, 512);

  // 2) per-head RMSNorm + RoPE on Q and K
  norm_rope_kernel<<<dim3(SEQ, NH + NKV), dim3(128), 0, stream>>>(
      qkv, cosb, sinb, qs, ks);

  // 3) causal GQA flash attention (bf16 in/out)
  attn_kernel<<<dim3(SEQ / 64, NH), dim3(128), 0, stream>>>(qkv, attO);

  // 4) output projection straight into d_out (bf16 in, f32 out)
  gemm_bf16_kernel<bf16, float><<<dim3(HIDDEN / 64, SEQ / 128), dim3(128), 0, stream>>>(
      attO, Wo, Wo, Wo, out, HIDDEN, HIDDEN, 0, 0);

  // 5) final RMSNorm in place
  final_norm_kernel<<<dim3(SEQ), dim3(256), 0, stream>>>(out, ls);
}
